// TransformerBlock_36721970380864
// MI455X (gfx1250) — matrix operand order repaired and checked by reading, not compiled, not run on hardware
//
#include <hip/hip_runtime.h>
#include <hip/hip_bf16.h>

typedef _Float16 f16;
typedef _Float16 v4h  __attribute__((ext_vector_type(4)));
typedef _Float16 v8h  __attribute__((ext_vector_type(8)));
typedef _Float16 v16h __attribute__((ext_vector_type(16)));
typedef float    v8f  __attribute__((ext_vector_type(8)));

#define LN_EPS 1e-5f
#define NSEQ   2048
#define DMODEL 1024
#define DH     64

__device__ __forceinline__ v8f zero8() {
  v8f z;
#pragma unroll
  for (int i = 0; i < 8; ++i) z[i] = 0.0f;
  return z;
}

// Build a 16-element f16 fragment from two contiguous 8-element LDS chunks.
__device__ __forceinline__ v16h make_frag(const f16* p0, const f16* p1) {
  v8h lo = *(const v8h*)p0;
  v8h hi = *(const v8h*)p1;
  v16h r;
#pragma unroll
  for (int i = 0; i < 8; ++i) { r[i] = lo[i]; r[i + 8] = hi[i]; }
  return r;
}

__device__ __forceinline__ v8f wmma_f16(v16h a, v16h b, v8f c) {
  return __builtin_amdgcn_wmma_f32_16x16x32_f16(false, a, false, b, (short)0, c,
                                                false, false);
}

// CDNA5 async global->LDS copy, 16 bytes per lane (ASYNCcnt tracked).
__device__ __forceinline__ void async_copy16(f16* lds, const f16* g) {
  uint32_t off = (uint32_t)(uintptr_t)lds;   // low 32 bits = LDS byte offset
  asm volatile("global_load_async_to_lds_b128 %0, %1, off"
               :: "v"(off), "v"(g)
               : "memory");
}
__device__ __forceinline__ void wait_async0() {
  asm volatile("s_wait_asynccnt 0" ::: "memory");
}

// ---------------------------------------------------------------- convert x
__global__ void __launch_bounds__(256) cvt_f32_f16(const float* __restrict__ in,
                                                   f16* __restrict__ out) {
  int idx = (blockIdx.x * 256 + threadIdx.x) * 4;
  float4 v = *(const float4*)(in + idx);
  v4h o;
  o[0] = (f16)v.x; o[1] = (f16)v.y; o[2] = (f16)v.z; o[3] = (f16)v.w;
  *(v4h*)(out + idx) = o;
}

// ------------------------------------------- transpose + convert weights
// in:  [R][C] f32 row-major      out: [C][R] f16 row-major (i.e. W^T)
__global__ void __launch_bounds__(256) transpose_cvt(const float* __restrict__ in,
                                                     f16* __restrict__ out,
                                                     int R, int C) {
  __shared__ f16 tile[32][33];
  const int tx = threadIdx.x & 31, ty = threadIdx.x >> 5;
  const int c0 = blockIdx.x * 32, r0 = blockIdx.y * 32;
#pragma unroll
  for (int i = 0; i < 4; ++i) {
    int r = ty + i * 8;
    tile[r][tx] = (f16)in[(size_t)(r0 + r) * C + c0 + tx];
  }
  __syncthreads();
#pragma unroll
  for (int i = 0; i < 4; ++i) {
    int r = ty + i * 8;
    out[(size_t)(c0 + r) * R + r0 + tx] = tile[tx][r];
  }
}

// ---------------------------------------------------------------- GEMM
// C[M][N] = A[M][K] * B[N][K]^T  (+bias, +epilogue), async double-buffered.
enum { EPI_H_BIAS = 0, EPI_H_GELU = 1, EPI_F32_RES = 2 };

template <int EPI>
__global__ void __launch_bounds__(256) gemm_f16(
    const f16* __restrict__ A, const f16* __restrict__ B,
    const float* __restrict__ bias, const float* __restrict__ resid,
    float* __restrict__ outF, f16* __restrict__ outH, int M, int N, int K) {
  __shared__ __align__(16) f16 As[2][128][32];
  __shared__ __align__(16) f16 Bs[2][128][32];

  const int tid  = threadIdx.x;
  const int lane = tid & 31, wave = tid >> 5;
  const int l15  = lane & 15, hi = (lane >> 4) & 1;
  const int wm   = wave >> 2, wn = wave & 3;   // 2x4 wave grid
  const int bm   = blockIdx.y * 128, bn = blockIdx.x * 128;

  v8f acc[4][2];
#pragma unroll
  for (int m = 0; m < 4; ++m)
#pragma unroll
    for (int n = 0; n < 2; ++n) acc[m][n] = zero8();

  const int lr = tid >> 1;           // tile row this thread stages
  const int lc = (tid & 1) * 16;     // tile col base (16 halfs)
  const f16* Aptr = A + (size_t)(bm + lr) * K + lc;
  const f16* Bptr = B + (size_t)(bn + lr) * K + lc;
  const int kIters = K >> 5;         // always even (32 or 128)

  auto issue = [&](int kt) {
    const int bi = kt & 1;
    async_copy16(&As[bi][lr][lc],     Aptr + kt * 32);
    async_copy16(&As[bi][lr][lc + 8], Aptr + kt * 32 + 8);
    async_copy16(&Bs[bi][lr][lc],     Bptr + kt * 32);
    async_copy16(&Bs[bi][lr][lc + 8], Bptr + kt * 32 + 8);
  };

  issue(0);
#pragma unroll 2
  for (int kt = 0; kt < kIters; ++kt) {
    wait_async0();       // own async copies for tile kt done
    __syncthreads();     // everyone's copies done; prev compute done
    if (kt + 1 < kIters) issue(kt + 1);   // overlap next copy with WMMAs
    const int bi = kt & 1;

    v16h bfrag[2];
#pragma unroll
    for (int n = 0; n < 2; ++n) {
      const f16* p = &Bs[bi][wn * 32 + n * 16 + l15][hi * 16];
      bfrag[n] = make_frag(p, p + 8);
    }
#pragma unroll
    for (int m = 0; m < 4; ++m) {
      const f16* p = &As[bi][wm * 64 + m * 16 + l15][hi * 16];
      v16h afrag = make_frag(p, p + 8);
#pragma unroll
      for (int n = 0; n < 2; ++n) acc[m][n] = wmma_f16(afrag, bfrag[n], acc[m][n]);
    }
  }

#pragma unroll
  for (int m = 0; m < 4; ++m) {
#pragma unroll
    for (int n = 0; n < 2; ++n) {
#pragma unroll
      for (int r = 0; r < 8; ++r) {
        int row = bm + wm * 64 + m * 16 + r + hi * 8;
        int col = bn + wn * 32 + n * 16 + l15;
        size_t o = (size_t)row * N + col;
        float v = acc[m][n][r] + bias[col];
        if (EPI == EPI_F32_RES) {
          outF[o] = v + resid[o];
        } else if (EPI == EPI_H_GELU) {
          v = 0.5f * v * (1.0f + erff(v * 0.70710678118654752f));
          outH[o] = (f16)v;
        } else {
          outH[o] = (f16)v;
        }
      }
    }
  }
}

// ----------------------------------------------------- flash attention
// qkv: [B*N][3*DMODEL] f16 (bias included).  out: [B*N][DMODEL] f16.
__global__ void __launch_bounds__(128) attention_k(const f16* __restrict__ qkv,
                                                   f16* __restrict__ outH) {
  __shared__ __align__(16) f16 Kt[64][64];   // [key][dh]
  __shared__ __align__(16) f16 Vt[64][64];   // [dh][key] (transposed)
  __shared__ __align__(16) f16 Pt[64][64];   // Q staging, then per-wave P tiles

  const int tid  = threadIdx.x;
  const int lane = tid & 31, wave = tid >> 5;
  const int l15  = lane & 15, hi = (lane >> 4) & 1;
  const int bh = blockIdx.x, qb = blockIdx.y;
  const int b = bh >> 4, h = bh & 15;
  const size_t rowb = (size_t)b * NSEQ;
  const int ld = 3 * DMODEL;

  // stage Q block (64 x 64) into Pt via async copy, build A-fragments
#pragma unroll
  for (int i = 0; i < 4; ++i) {
    int cc = tid + i * 128;
    int r = cc >> 3, d8 = (cc & 7) * 8;
    async_copy16(&Pt[r][d8], qkv + (rowb + qb * 64 + r) * ld + h * DH + d8);
  }
  wait_async0();
  __syncthreads();
  v16h qfrag[2];
#pragma unroll
  for (int g = 0; g < 2; ++g) {
    const f16* p = &Pt[wave * 16 + l15][g * 32 + hi * 16];
    qfrag[g] = make_frag(p, p + 8);
  }

  v8f oacc[4];
#pragma unroll
  for (int dj = 0; dj < 4; ++dj) oacc[dj] = zero8();
  float mstate[8], lstate[8];
#pragma unroll
  for (int r = 0; r < 8; ++r) { mstate[r] = -3e38f; lstate[r] = 0.0f; }

  for (int kb = 0; kb <= qb; ++kb) {
    __syncthreads();   // protect K/V (and Q staging on first iter) from overwrite
#pragma unroll
    for (int i = 0; i < 4; ++i) {
      int cc = tid + i * 128;
      int kr = cc >> 3, d8 = (cc & 7) * 8;
      async_copy16(&Kt[kr][d8],
                   qkv + (rowb + kb * 64 + kr) * ld + DMODEL + h * DH + d8);
      const f16* vs = qkv + (rowb + kb * 64 + kr) * ld + 2 * DMODEL + h * DH + d8;
      v8h vv = *(const v8h*)vs;
#pragma unroll
      for (int j = 0; j < 8; ++j) Vt[d8 + j][kr] = vv[j];
    }
    wait_async0();
    __syncthreads();

    // S = Q K^T   (4 key sub-tiles of 16)
    v8f sacc[4];
#pragma unroll
    for (int nj = 0; nj < 4; ++nj) {
      v8f s = zero8();
#pragma unroll
      for (int g = 0; g < 2; ++g) {
        const f16* p = &Kt[nj * 16 + l15][g * 32 + hi * 16];
        v16h bf = make_frag(p, p + 8);
        s = wmma_f16(qfrag[g], bf, s);
      }
      sacc[nj] = s;
    }
    const bool diag = (kb == qb);
#pragma unroll
    for (int nj = 0; nj < 4; ++nj) {
#pragma unroll
      for (int r = 0; r < 8; ++r) {
        float s = sacc[nj][r] * 0.125f;   // d_head^-0.5
        if (diag) {
          int q = wave * 16 + r + hi * 8;
          int k = nj * 16 + l15;
          if (k > q) s = -3e38f;
        }
        sacc[nj][r] = s;
      }
    }
    // online softmax (row lives in one 16-lane half)
#pragma unroll
    for (int r = 0; r < 8; ++r) {
      float rmax = -3e38f;
#pragma unroll
      for (int nj = 0; nj < 4; ++nj) rmax = fmaxf(rmax, sacc[nj][r]);
      rmax = fmaxf(rmax, __shfl_xor(rmax, 1, 32));
      rmax = fmaxf(rmax, __shfl_xor(rmax, 2, 32));
      rmax = fmaxf(rmax, __shfl_xor(rmax, 4, 32));
      rmax = fmaxf(rmax, __shfl_xor(rmax, 8, 32));
      float newm  = fmaxf(mstate[r], rmax);
      float alpha = expf(mstate[r] - newm);
      float rsum = 0.0f;
#pragma unroll
      for (int nj = 0; nj < 4; ++nj) {
        float p = expf(sacc[nj][r] - newm);
        rsum += p;
        Pt[wave * 16 + r + hi * 8][nj * 16 + l15] = (f16)p;
      }
      rsum += __shfl_xor(rsum, 1, 32);
      rsum += __shfl_xor(rsum, 2, 32);
      rsum += __shfl_xor(rsum, 4, 32);
      rsum += __shfl_xor(rsum, 8, 32);
      lstate[r] = lstate[r] * alpha + rsum;
      mstate[r] = newm;
#pragma unroll
      for (int dj = 0; dj < 4; ++dj) oacc[dj][r] *= alpha;
    }
    // O += P V (P read back from own wave's LDS rows; same-wave DS ordering)
#pragma unroll
    for (int g = 0; g < 2; ++g) {
      const f16* pp = &Pt[wave * 16 + l15][g * 32 + hi * 16];
      v16h pfrag = make_frag(pp, pp + 8);
#pragma unroll
      for (int dj = 0; dj < 4; ++dj) {
        const f16* vp = &Vt[dj * 16 + l15][g * 32 + hi * 16];
        v16h vfrag = make_frag(vp, vp + 8);
        oacc[dj] = wmma_f16(pfrag, vfrag, oacc[dj]);
      }
    }
  }

#pragma unroll
  for (int r = 0; r < 8; ++r) {
    float inv = 1.0f / lstate[r];
    int qrow = qb * 64 + wave * 16 + r + hi * 8;
#pragma unroll
    for (int dj = 0; dj < 4; ++dj) {
      int d = dj * 16 + l15;
      outH[(rowb + qrow) * DMODEL + h * DH + d] = (f16)(oacc[dj][r] * inv);
    }
  }
}

// ---------------------------------------------------------------- LayerNorm
template <bool WH>
__global__ void __launch_bounds__(256) layernorm_k(
    const float* __restrict__ in, const float* __restrict__ gam,
    const float* __restrict__ bet, float* __restrict__ outF,
    f16* __restrict__ outH) {
  __shared__ float ssum[8], ssq[8];
  const int row = blockIdx.x;
  const float* x = in + (size_t)row * DMODEL;
  const int c = threadIdx.x * 4;
  float4 v = *(const float4*)(x + c);
  float s = v.x + v.y + v.z + v.w;
  float q = v.x * v.x + v.y * v.y + v.z * v.z + v.w * v.w;
#pragma unroll
  for (int m = 1; m < 32; m <<= 1) {
    s += __shfl_xor(s, m, 32);
    q += __shfl_xor(q, m, 32);
  }
  if ((threadIdx.x & 31) == 0) { ssum[threadIdx.x >> 5] = s; ssq[threadIdx.x >> 5] = q; }
  __syncthreads();
  float S = 0.0f, Q = 0.0f;
#pragma unroll
  for (int i = 0; i < 8; ++i) { S += ssum[i]; Q += ssq[i]; }
  float mu  = S * (1.0f / DMODEL);
  float var = Q * (1.0f / DMODEL) - mu * mu;
  float rs  = rsqrtf(var + LN_EPS);
  float4 g = *(const float4*)(gam + c);
  float4 bb = *(const float4*)(bet + c);
  float4 o;
  o.x = (v.x - mu) * rs * g.x + bb.x;
  o.y = (v.y - mu) * rs * g.y + bb.y;
  o.z = (v.z - mu) * rs * g.z + bb.z;
  o.w = (v.w - mu) * rs * g.w + bb.w;
  *(float4*)(outF + (size_t)row * DMODEL + c) = o;
  if (WH) {
    v4h oh;
    oh[0] = (f16)o.x; oh[1] = (f16)o.y; oh[2] = (f16)o.z; oh[3] = (f16)o.w;
    *(v4h*)(outH + (size_t)row * DMODEL + c) = oh;
  }
}

// ---------------------------------------------------------------- launch
extern "C" void kernel_launch(void* const* d_in, const int* in_sizes, int n_in,
                              void* d_out, int out_size, void* d_ws, size_t ws_size,
                              hipStream_t stream) {
  const float* x     = (const float*)d_in[0];
  const float* w_qkv = (const float*)d_in[1];
  const float* b_qkv = (const float*)d_in[2];
  const float* w_out = (const float*)d_in[3];
  const float* b_out = (const float*)d_in[4];
  const float* ln1_g = (const float*)d_in[5];
  const float* ln1_b = (const float*)d_in[6];
  const float* w_ff1 = (const float*)d_in[7];
  const float* b_ff1 = (const float*)d_in[8];
  const float* w_ff2 = (const float*)d_in[9];
  const float* b_ff2 = (const float*)d_in[10];
  const float* ln2_g = (const float*)d_in[11];
  const float* ln2_b = (const float*)d_in[12];

  const size_t M = 4096;  // B*N = 2*2048
  char* wsp = (char*)d_ws;
  auto alloc = [&](size_t bytes) -> void* {
    void* p = (void*)wsp;
    wsp += (bytes + 255) & ~(size_t)255;
    return p;
  };
  f16*   xh     = (f16*)alloc(M * 1024 * 2);
  f16*   wqkvT  = (f16*)alloc(3072ull * 1024 * 2);
  f16*   woutT  = (f16*)alloc(1024ull * 1024 * 2);
  f16*   wff1T  = (f16*)alloc(4096ull * 1024 * 2);
  f16*   wff2T  = (f16*)alloc(1024ull * 4096 * 2);
  f16*   qkvh   = (f16*)alloc(M * 3072 * 2);
  f16*   attnh  = (f16*)alloc(M * 1024 * 2);
  float* resid1 = (float*)alloc(M * 1024 * 4);
  float* x1f    = (float*)alloc(M * 1024 * 4);
  f16*   x1h    = (f16*)alloc(M * 1024 * 2);
  f16*   hh     = (f16*)alloc(M * 4096 * 2);
  float* resid2 = (float*)alloc(M * 1024 * 4);

  cvt_f32_f16<<<(M * 1024 / 4) / 256, 256, 0, stream>>>(x, xh);
  transpose_cvt<<<dim3(3072 / 32, 1024 / 32), 256, 0, stream>>>(w_qkv, wqkvT, 1024, 3072);
  transpose_cvt<<<dim3(1024 / 32, 1024 / 32), 256, 0, stream>>>(w_out, woutT, 1024, 1024);
  transpose_cvt<<<dim3(4096 / 32, 1024 / 32), 256, 0, stream>>>(w_ff1, wff1T, 1024, 4096);
  transpose_cvt<<<dim3(1024 / 32, 4096 / 32), 256, 0, stream>>>(w_ff2, wff2T, 4096, 1024);

  // QKV projection
  gemm_f16<EPI_H_BIAS><<<dim3(3072 / 128, M / 128), 256, 0, stream>>>(
      xh, wqkvT, b_qkv, nullptr, nullptr, qkvh, (int)M, 3072, 1024);
  // causal flash attention
  attention_k<<<dim3(32, NSEQ / 64), 128, 0, stream>>>(qkvh, attnh);
  // out projection + residual
  gemm_f16<EPI_F32_RES><<<dim3(1024 / 128, M / 128), 256, 0, stream>>>(
      attnh, woutT, b_out, x, resid1, nullptr, (int)M, 1024, 1024);
  layernorm_k<true><<<M, 256, 0, stream>>>(resid1, ln1_g, ln1_b, x1f, x1h);
  // FFN
  gemm_f16<EPI_H_GELU><<<dim3(4096 / 128, M / 128), 256, 0, stream>>>(
      x1h, wff1T, b_ff1, nullptr, nullptr, hh, (int)M, 4096, 1024);
  gemm_f16<EPI_F32_RES><<<dim3(1024 / 128, M / 128), 256, 0, stream>>>(
      hh, wff2T, b_ff2, x1f, resid2, nullptr, (int)M, 1024, 4096);
  layernorm_k<false><<<M, 256, 0, stream>>>(resid2, ln2_g, ln2_b, (float*)d_out, nullptr);
}
